// MyCrossAttention_44650480009704
// MI455X (gfx1250) — matrix operand order repaired and checked by reading, not compiled, not run on hardware
//
#include <hip/hip_runtime.h>

#define BB   2
#define NN   1024
#define DD   1024
#define HH   16
#define DHD  64

typedef __attribute__((ext_vector_type(16))) __bf16 bf16x16;
typedef __attribute__((ext_vector_type(8)))  __bf16 bf16x8;
typedef __attribute__((ext_vector_type(8)))  float  f32x8;

__device__ __forceinline__ unsigned short f32_to_bf16(float f) {
  unsigned int u = __float_as_uint(f);
  u += 0x7fffu + ((u >> 16) & 1u);   // round to nearest even
  return (unsigned short)(u >> 16);
}

__device__ __forceinline__ bf16x16 cat8(bf16x8 lo, bf16x8 hi8) {
  bf16x16 f;
#pragma unroll
  for (int i = 0; i < 8; ++i) { f[i] = lo[i]; f[i + 8] = hi8[i]; }
  return f;
}

// A fragment: 16 rows x 32 K, row-major at `base` (row stride ld), K-chunk kc.
// ISA layout: lane<16 holds row r, K{kc..kc+7, kc+16..kc+23}; lane>=16 holds +8.
__device__ __forceinline__ bf16x16 ldA(const unsigned short* base, int ld, int kc,
                                       int r, int hi) {
  const unsigned short* p = base + (size_t)r * ld + kc + (hi ? 8 : 0);
  return cat8(*(const bf16x8*)p, *(const bf16x8*)(p + 16));
}

// B fragment from B^T storage: 16 rows of B^T (= output columns), row stride ld.
// lane<16 holds col r, K{kc..kc+7, kc+16..kc+23}; lane>=16 holds +8.
__device__ __forceinline__ bf16x16 ldB(const unsigned short* base, int ld, int kc,
                                       int r, int hi) {
  const unsigned short* p = base + (size_t)r * ld + kc + (hi ? 8 : 0);
  return cat8(*(const bf16x8*)p, *(const bf16x8*)(p + 16));
}

__device__ __forceinline__ f32x8 wmma_bf16(bf16x16 a, bf16x16 b, f32x8 c) {
  return __builtin_amdgcn_wmma_f32_16x16x32_bf16(false, a, false, b, (short)0, c,
                                                 false, false);
}

// ---------------------------------------------------------------- converts
__global__ __launch_bounds__(256) void f2bf_kernel(const float* __restrict__ in,
                                                   unsigned short* __restrict__ out,
                                                   int n) {
  int i = blockIdx.x * 256 + threadIdx.x;
  if (i < n) out[i] = f32_to_bf16(in[i]);
}

__global__ __launch_bounds__(256) void transpose_f32_to_bf16(
    const float* __restrict__ in, unsigned short* __restrict__ out,
    int rows, int cols) {
  __shared__ float tile[32][33];
  int x = blockIdx.x * 32 + threadIdx.x;
#pragma unroll
  for (int j = 0; j < 32; j += 8) {
    int yy = blockIdx.y * 32 + threadIdx.y + j;
    tile[threadIdx.y + j][threadIdx.x] = in[(size_t)yy * cols + x];
  }
  __syncthreads();
  int x2 = blockIdx.y * 32 + threadIdx.x;
#pragma unroll
  for (int j = 0; j < 32; j += 8) {
    int y2 = blockIdx.x * 32 + threadIdx.y + j;
    out[(size_t)y2 * rows + x2] = f32_to_bf16(tile[threadIdx.x][threadIdx.y + j]);
  }
}

__global__ __launch_bounds__(256) void transpose_bf16_kernel(
    const unsigned short* __restrict__ in, unsigned short* __restrict__ out,
    int rows, int cols) {
  __shared__ unsigned short tile[32][33];
  size_t bo = (size_t)blockIdx.z * rows * cols;
  int x = blockIdx.x * 32 + threadIdx.x;
#pragma unroll
  for (int j = 0; j < 32; j += 8) {
    int yy = blockIdx.y * 32 + threadIdx.y + j;
    tile[threadIdx.y + j][threadIdx.x] = in[bo + (size_t)yy * cols + x];
  }
  __syncthreads();
  int x2 = blockIdx.y * 32 + threadIdx.x;
#pragma unroll
  for (int j = 0; j < 32; j += 8) {
    int y2 = blockIdx.x * 32 + threadIdx.y + j;
    out[bo + (size_t)y2 * rows + x2] = tile[threadIdx.x][threadIdx.y + j];
  }
}

// ---------------------------------------------------------------- GEMM: C = A * Bt^T
// A: MxK bf16 row-major, Bt: NxK bf16 row-major (B transposed), C: MxN bf16.
// Wave = 32x32 tile; workgroup = 8 waves as 2(M) x 4(N) -> 64x128.
__global__ __launch_bounds__(256) void gemm_nt_kernel(
    const unsigned short* __restrict__ A, const unsigned short* __restrict__ Bt,
    unsigned short* __restrict__ C, int M, int N, int K, float scale) {
  const int lane = threadIdx.x & 31, wave = threadIdx.x >> 5;
  const int r = lane & 15, hi = lane >> 4;
  const int wm = wave >> 2, wn = wave & 3;
  const int m0 = blockIdx.y * 64 + wm * 32;
  const int n0 = blockIdx.x * 128 + wn * 32;
  const unsigned short* Ab = A + (size_t)m0 * K;
  const unsigned short* Bb = Bt + (size_t)n0 * K;
  f32x8 acc[2][2] = {};
  for (int kc = 0; kc < K; kc += 32) {
    __builtin_prefetch(Ab + (size_t)r * K + kc + 128, 0, 3);
    __builtin_prefetch(Bb + (size_t)r * K + kc + 128, 0, 3);
    bf16x16 a0 = ldA(Ab, K, kc, r, hi);
    bf16x16 a1 = ldA(Ab + (size_t)16 * K, K, kc, r, hi);
    bf16x16 b0 = ldB(Bb, K, kc, r, hi);
    bf16x16 b1 = ldB(Bb + (size_t)16 * K, K, kc, r, hi);
    acc[0][0] = wmma_bf16(a0, b0, acc[0][0]);
    acc[0][1] = wmma_bf16(a0, b1, acc[0][1]);
    acc[1][0] = wmma_bf16(a1, b0, acc[1][0]);
    acc[1][1] = wmma_bf16(a1, b1, acc[1][1]);
  }
#pragma unroll
  for (int i = 0; i < 2; ++i)
#pragma unroll
    for (int j = 0; j < 2; ++j)
#pragma unroll
      for (int v = 0; v < 8; ++v) {
        int row = m0 + i * 16 + hi * 8 + v;
        int col = n0 + j * 16 + r;
        C[(size_t)row * N + col] = f32_to_bf16(acc[i][j][v] * scale);
      }
}

// ---------------------------------------------------------------- t = SCALE^2 * q[b] @ k_h
// Per z=(b,h): A = q[b] (1024x1024), Bt = kT[b] rows h*64.. (64xK=1024), C = t[z] (1024x64).
// Workgroup = 8 waves as 4(M) x 2(N) -> 128x64.
__global__ __launch_bounds__(256) void gemm_t_kernel(
    const unsigned short* __restrict__ q, const unsigned short* __restrict__ kT,
    unsigned short* __restrict__ t) {
  const int lane = threadIdx.x & 31, wave = threadIdx.x >> 5;
  const int r = lane & 15, hi = lane >> 4;
  const int wm = wave >> 1, wn = wave & 1;
  const int z = blockIdx.y, b = z >> 4, h = z & 15;
  const int m0 = blockIdx.x * 128 + wm * 32;
  const int n0 = wn * 32;
  const unsigned short* Ab = q + (size_t)b * NN * DD + (size_t)m0 * DD;
  const unsigned short* Bb = kT + (size_t)b * DD * NN + (size_t)(h * DHD + n0) * NN;
  f32x8 acc[2][2] = {};
  for (int kc = 0; kc < DD; kc += 32) {
    bf16x16 a0 = ldA(Ab, DD, kc, r, hi);
    bf16x16 a1 = ldA(Ab + (size_t)16 * DD, DD, kc, r, hi);
    bf16x16 b0 = ldB(Bb, NN, kc, r, hi);
    bf16x16 b1 = ldB(Bb + (size_t)16 * NN, NN, kc, r, hi);
    acc[0][0] = wmma_bf16(a0, b0, acc[0][0]);
    acc[0][1] = wmma_bf16(a0, b1, acc[0][1]);
    acc[1][0] = wmma_bf16(a1, b0, acc[1][0]);
    acc[1][1] = wmma_bf16(a1, b1, acc[1][1]);
  }
  unsigned short* Cb = t + ((size_t)z * NN + m0) * DHD + n0;
  const float s2 = 1.0f / 64.0f;  // SCALE^2 = (dh^-0.5)^2
#pragma unroll
  for (int i = 0; i < 2; ++i)
#pragma unroll
    for (int j = 0; j < 2; ++j)
#pragma unroll
      for (int v = 0; v < 8; ++v)
        Cb[(size_t)(i * 16 + hi * 8 + v) * DHD + j * 16 + r] =
            f32_to_bf16(acc[i][j][v] * s2);
}

// ---------------------------------------------------------------- flash: softmax(t k_h^T) v_h
__global__ __launch_bounds__(256) void flash_kernel(
    const unsigned short* __restrict__ t, const unsigned short* __restrict__ kmat,
    const unsigned short* __restrict__ vT, float* __restrict__ attn_out) {
  const int lane = threadIdx.x & 31, wave = threadIdx.x >> 5;
  const int r = lane & 15, hi = lane >> 4;
  const int bh = blockIdx.y, b = bh >> 4, h = bh & 15;
  const int i0 = blockIdx.x * 256 + wave * 32;

  const unsigned short* tB = t + ((size_t)bh * NN + i0) * DHD;
  const unsigned short* kB = kmat + (size_t)b * NN * DD + h * DHD;       // rows j, cols e
  const unsigned short* vB = vT + (size_t)b * DD * NN + (size_t)h * DHD * NN;  // rows e, cols j

  bf16x16 at[2][2];
#pragma unroll
  for (int is = 0; is < 2; ++is)
#pragma unroll
    for (int kc = 0; kc < 2; ++kc)
      at[is][kc] = ldA(tB + (size_t)is * 16 * DHD, DHD, kc * 32, r, hi);

  f32x8 o[2][4] = {};
  float m[2][8], l[2][8];
#pragma unroll
  for (int is = 0; is < 2; ++is)
#pragma unroll
    for (int v = 0; v < 8; ++v) { m[is][v] = -3.0e38f; l[is][v] = 0.0f; }

  __shared__ __align__(16) unsigned short pbuf[8][32 * 32];
  unsigned short* pw = pbuf[wave];

  for (int j0 = 0; j0 < NN; j0 += 32) {
    // scores s = t @ k_j^T  (32x32), K = 64
    f32x8 s[2][2] = {};
#pragma unroll
    for (int kc = 0; kc < 2; ++kc) {
      bf16x16 b0 = ldB(kB + (size_t)j0 * DD, DD, kc * 32, r, hi);
      bf16x16 b1 = ldB(kB + (size_t)(j0 + 16) * DD, DD, kc * 32, r, hi);
      s[0][0] = wmma_bf16(at[0][kc], b0, s[0][0]);
      s[0][1] = wmma_bf16(at[0][kc], b1, s[0][1]);
      s[1][0] = wmma_bf16(at[1][kc], b0, s[1][0]);
      s[1][1] = wmma_bf16(at[1][kc], b1, s[1][1]);
    }
    // online softmax (rows live across the 16-lane halves; VGPR v -> row v(+8))
#pragma unroll
    for (int is = 0; is < 2; ++is) {
      f32x8 mx, rs;
#pragma unroll
      for (int v = 0; v < 8; ++v) mx[v] = fmaxf(s[is][0][v], s[is][1][v]);
#pragma unroll
      for (int d = 1; d < 16; d <<= 1)
#pragma unroll
        for (int v = 0; v < 8; ++v) mx[v] = fmaxf(mx[v], __shfl_xor(mx[v], d, 32));
#pragma unroll
      for (int v = 0; v < 8; ++v) {
        float mn = fmaxf(m[is][v], mx[v]);
        float alpha = __expf(m[is][v] - mn);
        m[is][v] = mn;
        l[is][v] *= alpha;
#pragma unroll
        for (int e = 0; e < 4; ++e) o[is][e][v] *= alpha;
        float p0 = __expf(s[is][0][v] - mn);
        float p1 = __expf(s[is][1][v] - mn);
        s[is][0][v] = p0; s[is][1][v] = p1;
        rs[v] = p0 + p1;
      }
#pragma unroll
      for (int d = 1; d < 16; d <<= 1)
#pragma unroll
        for (int v = 0; v < 8; ++v) rs[v] += __shfl_xor(rs[v], d, 32);
#pragma unroll
      for (int v = 0; v < 8; ++v) l[is][v] += rs[v];
      // spill P tile (C layout) to per-wave LDS as bf16
#pragma unroll
      for (int js = 0; js < 2; ++js)
#pragma unroll
        for (int v = 0; v < 8; ++v)
          pw[(size_t)(is * 16 + hi * 8 + v) * 32 + js * 16 + r] =
              f32_to_bf16(s[is][js][v]);
    }
    asm volatile("s_wait_dscnt 0x0" ::: "memory");  // DS RAW: C-layout -> A-layout reload
    bf16x16 pa0 = ldA(pw, 32, 0, r, hi);
    bf16x16 pa1 = ldA(pw + 16 * 32, 32, 0, r, hi);
#pragma unroll
    for (int es = 0; es < 4; ++es) {
      bf16x16 bv = ldB(vB + (size_t)es * 16 * NN, NN, j0, r, hi);
      o[0][es] = wmma_bf16(pa0, bv, o[0][es]);
      o[1][es] = wmma_bf16(pa1, bv, o[1][es]);
    }
  }

  float* outB = attn_out + ((size_t)b * NN + i0) * DD + h * DHD;
#pragma unroll
  for (int is = 0; is < 2; ++is)
#pragma unroll
    for (int v = 0; v < 8; ++v) {
      float inv = 1.0f / l[is][v];
      int row = is * 16 + hi * 8 + v;
#pragma unroll
      for (int es = 0; es < 4; ++es)
        outB[(size_t)row * DD + es * 16 + r] = o[is][es][v] * inv;
    }
}

// ---------------------------------------------------------------- residual + LayerNorm
__global__ __launch_bounds__(256) void ln_kernel(
    const float* __restrict__ attn, const float* __restrict__ y,
    const float* __restrict__ gamma, const float* __restrict__ beta,
    float* __restrict__ out) {
  const int row = blockIdx.x, tid = threadIdx.x;
  const float* a = attn + (size_t)row * DD;
  const float* yy = y + (size_t)row * DD;
  float vals[4];
  float s = 0.0f;
#pragma unroll
  for (int i = 0; i < 4; ++i) {
    int c = tid + i * 256;
    vals[i] = a[c] + yy[c];
    s += vals[i];
  }
  __shared__ float red[256];
  red[tid] = s; __syncthreads();
  for (int off = 128; off > 0; off >>= 1) {
    if (tid < off) red[tid] += red[tid + off];
    __syncthreads();
  }
  float mean = red[0] * (1.0f / DD);
  __syncthreads();
  float vs = 0.0f;
#pragma unroll
  for (int i = 0; i < 4; ++i) { float d = vals[i] - mean; vs += d * d; }
  red[tid] = vs; __syncthreads();
  for (int off = 128; off > 0; off >>= 1) {
    if (tid < off) red[tid] += red[tid + off];
    __syncthreads();
  }
  float inv = rsqrtf(red[0] * (1.0f / DD) + 1e-5f);
#pragma unroll
  for (int i = 0; i < 4; ++i) {
    int c = tid + i * 256;
    out[(size_t)row * DD + c] = (vals[i] - mean) * inv * gamma[c] + beta[c];
  }
}

// ---------------------------------------------------------------- launch
extern "C" void kernel_launch(void* const* d_in, const int* in_sizes, int n_in,
                              void* d_out, int out_size, void* d_ws, size_t ws_size,
                              hipStream_t stream) {
  (void)in_sizes; (void)n_in; (void)out_size; (void)ws_size;
  const float* x     = (const float*)d_in[0];
  const float* y     = (const float*)d_in[1];
  const float* Wq    = (const float*)d_in[2];
  const float* Wk    = (const float*)d_in[3];
  const float* Wv    = (const float*)d_in[4];
  const float* gamma = (const float*)d_in[5];
  const float* beta  = (const float*)d_in[6];
  float* out = (float*)d_out;

  size_t off = 0;
  auto carve = [&](size_t bytes) -> void* {
    void* p = (char*)d_ws + off;
    off += (bytes + 255) & ~(size_t)255;
    return p;
  };
  const size_t EL = (size_t)BB * NN * DD;  // 2M elements
  unsigned short* x_bf = (unsigned short*)carve(EL * 2);
  unsigned short* y_bf = (unsigned short*)carve(EL * 2);
  unsigned short* WqT  = (unsigned short*)carve((size_t)DD * DD * 2);
  unsigned short* WkT  = (unsigned short*)carve((size_t)DD * DD * 2);
  unsigned short* WvT  = (unsigned short*)carve((size_t)DD * DD * 2);
  unsigned short* q_bf = (unsigned short*)carve(EL * 2);
  unsigned short* k_bf = (unsigned short*)carve(EL * 2);
  unsigned short* v_bf = (unsigned short*)carve(EL * 2);
  unsigned short* kT   = (unsigned short*)carve(EL * 2);
  unsigned short* vT   = (unsigned short*)carve(EL * 2);
  unsigned short* t_bf = (unsigned short*)carve((size_t)BB * HH * NN * DHD * 2);
  float*          attn = (float*)carve(EL * 4);

  f2bf_kernel<<<(int)(EL / 256), 256, 0, stream>>>(x, x_bf, (int)EL);
  f2bf_kernel<<<(int)(EL / 256), 256, 0, stream>>>(y, y_bf, (int)EL);
  transpose_f32_to_bf16<<<dim3(32, 32), dim3(32, 8), 0, stream>>>(Wq, WqT, DD, DD);
  transpose_f32_to_bf16<<<dim3(32, 32), dim3(32, 8), 0, stream>>>(Wk, WkT, DD, DD);
  transpose_f32_to_bf16<<<dim3(32, 32), dim3(32, 8), 0, stream>>>(Wv, WvT, DD, DD);

  gemm_nt_kernel<<<dim3(8, 32), 256, 0, stream>>>(x_bf, WqT, q_bf, BB * NN, DD, DD, 1.0f);
  gemm_nt_kernel<<<dim3(8, 32), 256, 0, stream>>>(y_bf, WkT, k_bf, BB * NN, DD, DD, 1.0f);
  gemm_nt_kernel<<<dim3(8, 32), 256, 0, stream>>>(y_bf, WvT, v_bf, BB * NN, DD, DD, 1.0f);

  transpose_bf16_kernel<<<dim3(32, 32, BB), dim3(32, 8), 0, stream>>>(k_bf, kT, NN, DD);
  transpose_bf16_kernel<<<dim3(32, 32, BB), dim3(32, 8), 0, stream>>>(v_bf, vT, NN, DD);

  gemm_t_kernel<<<dim3(8, BB * HH), 256, 0, stream>>>(q_bf, kT, t_bf);
  flash_kernel<<<dim3(4, BB * HH), 256, 0, stream>>>(t_bf, k_bf, vT, attn);
  ln_kernel<<<BB * NN, 256, 0, stream>>>(attn, y, gamma, beta, out);
}
